// xLSTMActorCritic_910533067708
// MI455X (gfx1250) — matrix operand order repaired and checked by reading, not compiled, not run on hardware
//
#include <hip/hip_runtime.h>
#include <hip/hip_bf16.h>

#define HID 128
#define SEQ 512
#define BAT 64
#define DIN 64
#define NH 4
#define HDIM 32
#define NI 13

typedef __attribute__((ext_vector_type(16))) __bf16 v16bf;
typedef __attribute__((ext_vector_type(8)))  float  v8f;

union UA { v16bf v; unsigned u[8]; };

__device__ __forceinline__ unsigned f2bf1(float f) {
  return (unsigned)__builtin_bit_cast(unsigned short, (__bf16)f);
}
__device__ __forceinline__ unsigned f2bfpk(float lo, float hi) {
  return f2bf1(lo) | (f2bf1(hi) << 16);
}
__device__ __forceinline__ v8f wmma_bf16(const UA& a, const UA& b, v8f c) {
  return __builtin_amdgcn_wmma_f32_16x16x32_bf16(false, a.v, false, b.v,
                                                 (short)0, c, false, false);
}
__device__ __forceinline__ void set8(UA& a, uint4 q0, uint4 q1) {
  a.u[0] = q0.x; a.u[1] = q0.y; a.u[2] = q0.z; a.u[3] = q0.w;
  a.u[4] = q1.x; a.u[5] = q1.y; a.u[6] = q1.z; a.u[7] = q1.w;
}
#define V8F_ZERO {0.f,0.f,0.f,0.f,0.f,0.f,0.f,0.f}

// ---- Pre-swizzle fp32 weights into bf16 WMMA B-fragment order ----
// dst[((nt*kchunks+kc)*32+lane)*8+v] = pack(W[nt*16+(lane&15)][off+k], ..k+1)
// with k = kc*32 + (v>>2)*16 + (lane>>4)*8 + 2(v&3)
__global__ void k_swizzle(const float* __restrict__ W, unsigned* __restrict__ dst,
                          int ntiles, int kchunks, int stride, int coloff) {
  int total = ntiles * kchunks * 256;
  for (int i = blockIdx.x * blockDim.x + threadIdx.x; i < total;
       i += gridDim.x * blockDim.x) {
    int v = i & 7, lane = (i >> 3) & 31, rest = i >> 8;
    int kc = rest % kchunks, nt = rest / kchunks;
    int row = nt * 16 + (lane & 15);
    int k = kc * 32 + (v >> 2) * 16 + (lane >> 4) * 8 + 2 * (v & 3);
    const float* p = W + (size_t)row * stride + coloff + k;
    dst[i] = f2bfpk(p[0], p[1]);
  }
}

// ---- Kernel 1: xproj[r][g] = x[r,:] . Wcell[g,0:64] + b_cell[g] ----
__global__ void __launch_bounds__(256) k_xproj(const float* __restrict__ x,
    const unsigned* __restrict__ wcx, const float* __restrict__ bcell,
    float* __restrict__ xproj) {
  int wid = (blockIdx.x * blockDim.x + threadIdx.x) >> 5;
  int lane = threadIdx.x & 31, lo = lane & 15, hi = lane >> 4;
  int mtile = wid >> 5, ntile = wid & 31;
  int row = mtile * 16 + lo;
  const float4* px = (const float4*)(x + (size_t)row * DIN);
  v8f acc = V8F_ZERO;
  for (int kc = 0; kc < 2; ++kc) {
    float4 f0 = px[kc * 8 + hi * 2];
    float4 f1 = px[kc * 8 + hi * 2 + 1];
    float4 f2 = px[kc * 8 + 4 + hi * 2];
    float4 f3 = px[kc * 8 + 4 + hi * 2 + 1];
    UA a, b;
    a.u[0] = f2bfpk(f0.x, f0.y); a.u[1] = f2bfpk(f0.z, f0.w);
    a.u[2] = f2bfpk(f1.x, f1.y); a.u[3] = f2bfpk(f1.z, f1.w);
    a.u[4] = f2bfpk(f2.x, f2.y); a.u[5] = f2bfpk(f2.z, f2.w);
    a.u[6] = f2bfpk(f3.x, f3.y); a.u[7] = f2bfpk(f3.z, f3.w);
    const uint4* wp = (const uint4*)(wcx + (((ntile * 2 + kc) * 32 + lane) << 3));
    set8(b, wp[0], wp[1]);
    acc = wmma_bf16(a, b, acc);
  }
#pragma unroll
  for (int v = 0; v < 8; ++v) {
    int m = v + hi * 8, g = ntile * 16 + lo;
    int r = mtile * 16 + m;
    xproj[(size_t)r * 512 + g] = acc[v] + bcell[g];
  }
}

// ---- Kernel 2: persistent LSTM scan, W_h fully register-resident ----
__global__ void __launch_bounds__(256, 2) k_scan(const float* __restrict__ xproj,
    const unsigned* __restrict__ whsw, float* __restrict__ seqh_f,
    unsigned short* __restrict__ seqh_b) {
  __shared__ __align__(16) float gates[16 * 512];
  __shared__ __align__(16) unsigned short hbuf[16 * 128];
  int tid = threadIdx.x;
  int wv = tid >> 5, lane = tid & 31, lo = lane & 15, hi = lane >> 4;
  int bbase = blockIdx.x * 16;

  // Preload this wave's 16 B-fragments (16 KB/lane-set) into registers.
  UA bfr[16];
#pragma unroll
  for (int f = 0; f < 16; ++f) {
    int nt = wv * 4 + (f >> 2), kc = f & 3;
    const uint4* wp = (const uint4*)(whsw + (((nt * 4 + kc) * 32 + lane) << 3));
    set8(bfr[f], wp[0], wp[1]);
  }

  float cr[8], nr[8];
#pragma unroll
  for (int e = 0; e < 8; ++e) { cr[e] = 0.f; nr[e] = 1.f; }
  ((uint4*)hbuf)[tid] = make_uint4(0u, 0u, 0u, 0u);
  __syncthreads();

  int em = tid >> 4;                // elementwise row 0..15
  int ej = (tid & 15) * 8;          // elementwise col base
  const uint4* hb4 = (const uint4*)hbuf;

  for (int t = 0; t < SEQ; ++t) {
    v8f acc[4] = {V8F_ZERO, V8F_ZERO, V8F_ZERO, V8F_ZERO};
#pragma unroll
    for (int kc = 0; kc < 4; ++kc) {
      UA afr;
      set8(afr, hb4[lo * 16 + kc * 4 + hi], hb4[lo * 16 + kc * 4 + 2 + hi]);
#pragma unroll
      for (int tn = 0; tn < 4; ++tn)
        acc[tn] = wmma_bf16(afr, bfr[tn * 4 + kc], acc[tn]);
    }
#pragma unroll
    for (int tn = 0; tn < 4; ++tn)
#pragma unroll
      for (int v = 0; v < 8; ++v)
        gates[(v + hi * 8) * 512 + (wv * 4 + tn) * 16 + lo] = acc[tn][v];
    __syncthreads();

    // Elementwise gate update: 8 consecutive units per thread.
    const float* xp = xproj + (size_t)((bbase + em) * SEQ + t) * 512;
    const float4* g4 = (const float4*)gates;
    int gb = em * 512 + ej;
    float4 i0 = g4[(gb      ) >> 2], i1 = g4[((gb      ) >> 2) + 1];
    float4 f0 = g4[(gb + 128) >> 2], f1 = g4[((gb + 128) >> 2) + 1];
    float4 o0 = g4[(gb + 256) >> 2], f2o1 = g4[((gb + 256) >> 2) + 1];
    float4 z0 = g4[(gb + 384) >> 2], z1 = g4[((gb + 384) >> 2) + 1];
    const float4* xp4 = (const float4*)(xp + ej);
    const float4* xf4 = (const float4*)(xp + 128 + ej);
    const float4* xo4 = (const float4*)(xp + 256 + ej);
    const float4* xz4 = (const float4*)(xp + 384 + ej);
    float gi[8], gf[8], go[8], gz[8];
    { float4 a = xp4[0], b = xp4[1];
      gi[0]=i0.x+a.x; gi[1]=i0.y+a.y; gi[2]=i0.z+a.z; gi[3]=i0.w+a.w;
      gi[4]=i1.x+b.x; gi[5]=i1.y+b.y; gi[6]=i1.z+b.z; gi[7]=i1.w+b.w; }
    { float4 a = xf4[0], b = xf4[1];
      gf[0]=f0.x+a.x; gf[1]=f0.y+a.y; gf[2]=f0.z+a.z; gf[3]=f0.w+a.w;
      gf[4]=f1.x+b.x; gf[5]=f1.y+b.y; gf[6]=f1.z+b.z; gf[7]=f1.w+b.w; }
    { float4 a = xo4[0], b = xo4[1];
      go[0]=o0.x+a.x; go[1]=o0.y+a.y; go[2]=o0.z+a.z; go[3]=o0.w+a.w;
      go[4]=f2o1.x+b.x; go[5]=f2o1.y+b.y; go[6]=f2o1.z+b.z; go[7]=f2o1.w+b.w; }
    { float4 a = xz4[0], b = xz4[1];
      gz[0]=z0.x+a.x; gz[1]=z0.y+a.y; gz[2]=z0.z+a.z; gz[3]=z0.w+a.w;
      gz[4]=z1.x+b.x; gz[5]=z1.y+b.y; gz[6]=z1.z+b.z; gz[7]=z1.w+b.w; }

    if (t + 1 < SEQ) {  // prefetch next step's gate pre-activations
      const float* nx = xproj + (size_t)((bbase + em) * SEQ + t + 1) * 512;
      __builtin_prefetch(nx + ej, 0, 1);
      __builtin_prefetch(nx + 128 + ej, 0, 1);
      __builtin_prefetch(nx + 256 + ej, 0, 1);
      __builtin_prefetch(nx + 384 + ej, 0, 1);
    }

    float hv[8];
#pragma unroll
    for (int e = 0; e < 8; ++e) {
      float I = __expf(fminf(fmaxf(gi[e], -5.f), 5.f));
      float F = __expf(fminf(fmaxf(gf[e], -5.f), 5.f));
      cr[e] = fminf(fmaxf(F * cr[e] + I * tanhf(gz[e]), -1e6f), 1e6f);
      nr[e] = fminf(fmaxf(F * nr[e] + I, 1e-6f), 1e6f);
      float h = (1.f / (1.f + __expf(-go[e]))) * (cr[e] / nr[e]);
      if (!__builtin_isfinite(h)) h = 0.f;
      hv[e] = h;
    }
    size_t r = (size_t)(bbase + em) * SEQ + t;
    float4* sf = (float4*)(seqh_f + r * HID + ej);
    sf[0] = make_float4(hv[0], hv[1], hv[2], hv[3]);
    sf[1] = make_float4(hv[4], hv[5], hv[6], hv[7]);
    uint4 hp = make_uint4(f2bfpk(hv[0], hv[1]), f2bfpk(hv[2], hv[3]),
                          f2bfpk(hv[4], hv[5]), f2bfpk(hv[6], hv[7]));
    *(uint4*)(seqh_b + r * HID + ej) = hp;
    __syncthreads();   // all h-fragment reads of step t done before update
    *(uint4*)(hbuf + em * 128 + ej) = hp;
    __syncthreads();
  }
}

// ---- Kernel 3: QKV projection, write q/k bf16 and v transposed bf16 ----
__global__ void __launch_bounds__(256) k_qkv(const unsigned short* __restrict__ seqh_b,
    const unsigned* __restrict__ wqsw, const float* __restrict__ bqkv,
    unsigned short* __restrict__ qb, unsigned short* __restrict__ kb,
    unsigned short* __restrict__ vtb) {
  int wid = (blockIdx.x * blockDim.x + threadIdx.x) >> 5;
  int lane = threadIdx.x & 31, lo = lane & 15, hi = lane >> 4;
  int mtile = wid / 24, ntile = wid % 24;
  int row = mtile * 16 + lo;
  const uint4* sp = (const uint4*)(seqh_b + (size_t)row * HID);
  v8f acc = V8F_ZERO;
  for (int kc = 0; kc < 4; ++kc) {
    UA a, b;
    set8(a, sp[kc * 4 + hi], sp[kc * 4 + 2 + hi]);
    const uint4* wp = (const uint4*)(wqsw + (((ntile * 4 + kc) * 32 + lane) << 3));
    set8(b, wp[0], wp[1]);
    acc = wmma_bf16(a, b, acc);
  }
#pragma unroll
  for (int v = 0; v < 8; ++v) {
    int m = v + hi * 8, col = ntile * 16 + lo;
    int r = mtile * 16 + m, bb = r >> 9, s = r & 511;
    unsigned short bv = (unsigned short)f2bf1(acc[v] + bqkv[col]);
    if (col < 128) {
      int hh = col >> 5, hd = col & 31;
      qb[((size_t)(bb * NH + hh) * SEQ + s) * HDIM + hd] = bv;
    } else if (col < 256) {
      int cc = col - 128, hh = cc >> 5, hd = cc & 31;
      kb[((size_t)(bb * NH + hh) * SEQ + s) * HDIM + hd] = bv;
    } else {
      int cc = col - 256, hh = cc >> 5, hd = cc & 31;
      vtb[((size_t)(bb * NH + hh) * HDIM + hd) * SEQ + s] = bv;
    }
  }
}

// ---- Kernel 4: attention, one wave per (b, head, 16-query tile) ----
__global__ void __launch_bounds__(32) k_attn(const unsigned short* __restrict__ qb,
    const unsigned short* __restrict__ kb, const unsigned short* __restrict__ vtb,
    unsigned short* __restrict__ avb) {
  __shared__ __align__(16) float sc[16 * 512];
  __shared__ __align__(16) unsigned short pr[16 * 512];
  __shared__ float inv[16];
  int lane = threadIdx.x, lo = lane & 15, hi = lane >> 4;
  int qt = blockIdx.x & 31, hh = (blockIdx.x >> 5) & 3, bb = blockIdx.x >> 7;
  int bh = bb * NH + hh;
  const uint4* q4 = (const uint4*)qb;
  const uint4* k4 = (const uint4*)kb;
  const uint4* v4 = (const uint4*)vtb;
  const uint4* p4 = (const uint4*)pr;
  UA qa;
  {
    size_t base = ((size_t)bh * SEQ + qt * 16 + lo) * 4;
    set8(qa, q4[base + hi], q4[base + 2 + hi]);
  }
  for (int kt = 0; kt < 32; ++kt) {
    UA b;
    size_t base = ((size_t)bh * SEQ + kt * 16 + lo) * 4;
    set8(b, k4[base + hi], k4[base + 2 + hi]);
    v8f acc = V8F_ZERO;
    acc = wmma_bf16(qa, b, acc);
#pragma unroll
    for (int v = 0; v < 8; ++v)
      sc[(v + hi * 8) * 512 + kt * 16 + lo] = acc[v];
  }
  __syncthreads();
  const float scale = 0.17677669529663687f; // 1/sqrt(32)
  const float4* s4 = (const float4*)sc;
  int rbase4 = lo * 128 + hi * 64;          // float4 index of row half
  float mx = -3.4e38f;
  for (int i = 0; i < 64; ++i) {
    float4 s = s4[rbase4 + i];
    mx = fmaxf(mx, fmaxf(fmaxf(s.x, s.y), fmaxf(s.z, s.w)));
  }
  mx = fmaxf(mx, __shfl_xor(mx, 16, 32));
  mx *= scale;
  float sum = 0.f;
  uint2* pr2 = (uint2*)pr;
  for (int i = 0; i < 64; ++i) {
    float4 s = s4[rbase4 + i];
    float e0 = __expf(s.x * scale - mx), e1 = __expf(s.y * scale - mx);
    float e2 = __expf(s.z * scale - mx), e3 = __expf(s.w * scale - mx);
    sum += (e0 + e1) + (e2 + e3);
    pr2[rbase4 + i] = make_uint2(f2bfpk(e0, e1), f2bfpk(e2, e3));
  }
  sum += __shfl_xor(sum, 16, 32);
  if (hi == 0) inv[lo] = 1.f / sum;
  __syncthreads();
  v8f a0 = V8F_ZERO, a1 = V8F_ZERO;
  size_t vrow0 = (size_t)(bh * HDIM + lo) * 64;
  size_t vrow1 = (size_t)(bh * HDIM + 16 + lo) * 64;
  for (int kc = 0; kc < 16; ++kc) {
    UA a, b0, b1;
    set8(a, p4[lo * 64 + kc * 4 + hi], p4[lo * 64 + kc * 4 + 2 + hi]);
    set8(b0, v4[vrow0 + kc * 4 + hi], v4[vrow0 + kc * 4 + 2 + hi]);
    set8(b1, v4[vrow1 + kc * 4 + hi], v4[vrow1 + kc * 4 + 2 + hi]);
    a0 = wmma_bf16(a, b0, a0);
    a1 = wmma_bf16(a, b1, a1);
  }
#pragma unroll
  for (int v = 0; v < 8; ++v) {
    int m = v + hi * 8;
    size_t r = (size_t)bb * SEQ + qt * 16 + m;
    float is = inv[m];
    avb[r * HID + hh * HDIM + lo]      = (unsigned short)f2bf1(a0[v] * is);
    avb[r * HID + hh * HDIM + 16 + lo] = (unsigned short)f2bf1(a1[v] * is);
  }
}

// ---- Kernel 5: out-proj + residual + LayerNorm ----
__global__ void __launch_bounds__(256) k_oproj(const unsigned short* __restrict__ avb,
    const unsigned* __restrict__ wosw, const float* __restrict__ bo,
    const float* __restrict__ seqh_f, const float* __restrict__ lng,
    const float* __restrict__ lnb, float* __restrict__ lnout) {
  __shared__ __align__(16) float y[16 * 128];
  __shared__ float mu[16], rs[16];
  int tid = threadIdx.x, wv = tid >> 5, lane = tid & 31, lo = lane & 15, hi = lane >> 4;
  int mtile = blockIdx.x;
  int row = mtile * 16 + lo;
  const uint4* ap = (const uint4*)(avb + (size_t)row * HID);
  v8f acc = V8F_ZERO;
  for (int kc = 0; kc < 4; ++kc) {
    UA a, b;
    set8(a, ap[kc * 4 + hi], ap[kc * 4 + 2 + hi]);
    const uint4* wp = (const uint4*)(wosw + (((wv * 4 + kc) * 32 + lane) << 3));
    set8(b, wp[0], wp[1]);
    acc = wmma_bf16(a, b, acc);
  }
#pragma unroll
  for (int v = 0; v < 8; ++v) {
    int m = v + hi * 8, col = wv * 16 + lo;
    int r = mtile * 16 + m;
    y[m * 128 + col] = acc[v] + bo[col] + seqh_f[(size_t)r * HID + col];
  }
  __syncthreads();
  if (tid < 16) {
    float s = 0.f;
    for (int j = 0; j < 128; ++j) s += y[tid * 128 + j];
    s *= (1.f / 128.f);
    float s2 = 0.f;
    for (int j = 0; j < 128; ++j) { float d = y[tid * 128 + j] - s; s2 += d * d; }
    mu[tid] = s;
    rs[tid] = rsqrtf(s2 * (1.f / 128.f) + 1e-5f);
  }
  __syncthreads();
#pragma unroll
  for (int e = 0; e < 8; ++e) {
    int u = tid * 8 + e, m = u >> 7, j = u & 127;
    int r = mtile * 16 + m;
    lnout[(size_t)r * HID + j] = lng[j] * (y[m * 128 + j] - mu[m]) * rs[m] + lnb[j];
  }
}

// ---- Kernel 5b: mean over sequence ----
__global__ void k_ctx(const float* __restrict__ lnout, float* __restrict__ ctx) {
  int b = blockIdx.x, j = threadIdx.x;
  float s = 0.f;
  for (int t = 0; t < SEQ; ++t) s += lnout[((size_t)b * SEQ + t) * HID + j];
  ctx[b * HID + j] = s * (1.f / SEQ);
}

// ---- Kernel 6: actor/critic heads ----
__global__ void __launch_bounds__(128) k_heads(const float* __restrict__ ctx,
    const float* __restrict__ info, const float* __restrict__ Wa1,
    const float* __restrict__ ba1, const float* __restrict__ lnag,
    const float* __restrict__ lnab, const float* __restrict__ Wa2,
    const float* __restrict__ ba2, const float* __restrict__ Wc1,
    const float* __restrict__ bc1, const float* __restrict__ lncg,
    const float* __restrict__ lncb, const float* __restrict__ Wc2,
    const float* __restrict__ bc2, float* __restrict__ out) {
  __shared__ float comb[HID + NI];
  __shared__ float a1[128], c1[128];
  __shared__ float st[4];
  __shared__ float logits[3];
  int b = blockIdx.x, j = threadIdx.x;
  comb[j] = ctx[b * HID + j];
  if (j < NI) comb[HID + j] = info[b * NI + j];
  __syncthreads();
  float sa = ba1[j], sc2 = bc1[j];
  for (int k = 0; k < HID + NI; ++k) {
    float cv = comb[k];
    sa += Wa1[j * (HID + NI) + k] * cv;
    sc2 += Wc1[j * (HID + NI) + k] * cv;
  }
  a1[j] = sa; c1[j] = sc2;
  __syncthreads();
  if (j < 2) {
    const float* src = (j == 0) ? a1 : c1;
    float m = 0.f;
    for (int k = 0; k < 128; ++k) m += src[k];
    m *= (1.f / 128.f);
    float v = 0.f;
    for (int k = 0; k < 128; ++k) { float d = src[k] - m; v += d * d; }
    st[j * 2] = m;
    st[j * 2 + 1] = rsqrtf(v * (1.f / 128.f) + 1e-5f);
  }
  __syncthreads();
  float ra = fmaxf(0.f, lnag[j] * (a1[j] - st[0]) * st[1] + lnab[j]);
  float rc = fmaxf(0.f, lncg[j] * (c1[j] - st[2]) * st[3] + lncb[j]);
  __syncthreads();
  a1[j] = ra; c1[j] = rc;
  __syncthreads();
  if (j < 3) {
    float l = ba2[j];
    for (int k = 0; k < 128; ++k) l += Wa2[j * 128 + k] * a1[k];
    logits[j] = l;
  }
  if (j == 3) {
    float v = bc2[0];
    for (int k = 0; k < 128; ++k) v += Wc2[k] * c1[k];
    out[BAT * 3 + b] = v;
  }
  __syncthreads();
  if (j == 0) {
    float m = fmaxf(logits[0], fmaxf(logits[1], logits[2]));
    float e0 = __expf(logits[0] - m), e1 = __expf(logits[1] - m),
          e2 = __expf(logits[2] - m);
    float is = 1.f / (e0 + e1 + e2);
    out[b * 3 + 0] = e0 * is;
    out[b * 3 + 1] = e1 * is;
    out[b * 3 + 2] = e2 * is;
  }
}

extern "C" void kernel_launch(void* const* d_in, const int* in_sizes, int n_in,
                              void* d_out, int out_size, void* d_ws, size_t ws_size,
                              hipStream_t stream) {
  (void)in_sizes; (void)n_in; (void)out_size; (void)ws_size;
  const float* x     = (const float*)d_in[0];
  const float* info  = (const float*)d_in[1];
  const float* Wcell = (const float*)d_in[2];
  const float* bcell = (const float*)d_in[3];
  const float* Wqkv  = (const float*)d_in[4];
  const float* bqkv  = (const float*)d_in[5];
  const float* Wo    = (const float*)d_in[6];
  const float* bo    = (const float*)d_in[7];
  const float* lng   = (const float*)d_in[8];
  const float* lnb   = (const float*)d_in[9];
  const float* Wa1   = (const float*)d_in[10];
  const float* ba1   = (const float*)d_in[11];
  const float* lnag  = (const float*)d_in[12];
  const float* lnab  = (const float*)d_in[13];
  const float* Wa2   = (const float*)d_in[14];
  const float* ba2   = (const float*)d_in[15];
  const float* Wc1   = (const float*)d_in[16];
  const float* bc1   = (const float*)d_in[17];
  const float* lncg  = (const float*)d_in[18];
  const float* lncb  = (const float*)d_in[19];
  const float* Wc2   = (const float*)d_in[20];
  const float* bc2   = (const float*)d_in[21];

  char* ws = (char*)d_ws;
  const size_t MB = 1ull << 20;
  float*          xproj = (float*)(ws);                 // 64MB, dead after scan
  float*          lnout = (float*)(ws);                 // reuses xproj region
  unsigned short* qb    = (unsigned short*)(ws + 16 * MB);
  unsigned short* kb    = (unsigned short*)(ws + 24 * MB);
  unsigned short* vtb   = (unsigned short*)(ws + 32 * MB);
  unsigned short* avb   = (unsigned short*)(ws + 40 * MB);
  float*          ctx   = (float*)(ws + 48 * MB);
  float*          seqhf = (float*)(ws + 64 * MB);
  unsigned short* seqhb = (unsigned short*)(ws + 80 * MB);
  unsigned*       wcx   = (unsigned*)(ws + 88 * MB);
  unsigned*       whsw  = (unsigned*)(ws + 88 * MB + 64 * 1024);
  unsigned*       wqsw  = (unsigned*)(ws + 88 * MB + 192 * 1024);
  unsigned*       wosw  = (unsigned*)(ws + 88 * MB + 288 * 1024);

  k_swizzle<<<64, 256, 0, stream>>>(Wcell, wcx, 32, 2, 192, 0);
  k_swizzle<<<128, 256, 0, stream>>>(Wcell, whsw, 32, 4, 192, 64);
  k_swizzle<<<96, 256, 0, stream>>>(Wqkv, wqsw, 24, 4, 128, 0);
  k_swizzle<<<32, 256, 0, stream>>>(Wo, wosw, 8, 4, 128, 0);

  k_xproj<<<8192, 256, 0, stream>>>(x, wcx, bcell, xproj);
  k_scan<<<4, 256, 0, stream>>>(xproj, whsw, seqhf, seqhb);
  k_qkv<<<6144, 256, 0, stream>>>(seqhb, wqsw, bqkv, qb, kb, vtb);
  k_attn<<<8192, 32, 0, stream>>>(qb, kb, vtb, avb);
  k_oproj<<<2048, 256, 0, stream>>>(avb, wosw, bo, seqhf, lng, lnb, lnout);
  k_ctx<<<64, 128, 0, stream>>>(lnout, ctx);
  k_heads<<<64, 128, 0, stream>>>(ctx, info, Wa1, ba1, lnag, lnab, Wa2, ba2,
                                  Wc1, bc1, lncg, lncb, Wc2, bc2, (float*)d_out);
}
